// FNOBlock1D_30090540876184
// MI455X (gfx1250) — compile-verified, matrix operand order checked by reading, not run on hardware
//
#include <hip/hip_runtime.h>
#include <hip/hip_bf16.h>

// FNO block: B=64, N=16384, C=32, MODES=16, DEPTH=4 — truncated-DFT-as-matmul on WMMA,
// with TDM (tensor_load_to_lds) double-buffered staging of x tiles in the projection.
typedef __attribute__((ext_vector_type(16))) __bf16 v16bf;
typedef __attribute__((ext_vector_type(8)))  float  v8f;
typedef unsigned int v4u  __attribute__((ext_vector_type(4)));
typedef int          v8i_ __attribute__((ext_vector_type(8)));
typedef int          v4i_ __attribute__((ext_vector_type(4)));

#define Bn      64
#define Nn      16384
#define Cn      32
#define MODESn  16
#define DEPTHn  4
#define NSPLIT  16
#define TILES_PER_WAVE 16
#define LDS_ROW 33   // 32 dwords + 1 pad dword (TDM pad feature) -> conflict-free columns

#define HAS_TDM (__has_builtin(__builtin_amdgcn_tensor_load_to_lds) && \
                 __has_builtin(__builtin_amdgcn_s_wait_tensorcnt))

static __device__ __forceinline__ float gelu_tanh(float h) {
    const float k0 = 0.79788456080286535588f;           // sqrt(2/pi)
    float u = k0 * (h + 0.044715f * h * h * h);
    return 0.5f * h * (1.0f + tanhf(u));
}

// ---------------------------------------------------------------------------
// Kernel 0: bf16 DFT basis. Ft[32][N] (transposed, forward A-operand),
// F[N][32] (row-major, inverse A-operand). Rows 0..15 = cos, 16..31 = sin.
// Exact phase wrap via (m*n) & (N-1), N = 2^14.
// ---------------------------------------------------------------------------
__global__ void __launch_bounds__(256) build_basis(__bf16* __restrict__ Ft,
                                                   __bf16* __restrict__ F) {
    int n = blockIdx.x * blockDim.x + threadIdx.x;
    if (n >= Nn) return;
    const float w = 6.28318530717958647692f / (float)Nn;
    #pragma unroll
    for (int m = 0; m < MODESn; ++m) {
        int ph = (m * n) & (Nn - 1);
        float s, c;
        __sincosf(w * (float)ph, &s, &c);
        F[n * (2 * MODESn) + m]          = (__bf16)c;
        F[n * (2 * MODESn) + MODESn + m] = (__bf16)s;
        Ft[(size_t)m * Nn + n]            = (__bf16)c;
        Ft[(size_t)(MODESn + m) * Nn + n] = (__bf16)s;
    }
}

// ---------------------------------------------------------------------------
// Kernel 1: forward projection XF = Ft(32xN) * x[b](Nx32), split-K over NSPLIT.
// One wave per WG owns the whole 32x32 output (2x2 WMMA tiles). x tiles are
// staged to LDS by the Tensor Data Mover (double-buffered, padded rows), the
// wave reads bank-conflict-free columns with ds_load_b32 and converts to bf16.
// f32 partial per (b, split) -> ws (order-fixed reduction, no float atomics).
// ---------------------------------------------------------------------------
__global__ void __launch_bounds__(32) fno_project(const float*  __restrict__ x,
                                                  const __bf16* __restrict__ Ft,
                                                  float*        __restrict__ partial) {
    const int b    = blockIdx.x;
    const int s    = blockIdx.y;
    const int lane = threadIdx.x;
    const int lo   = lane & 15;     // A: M row / B: N column / D: N column
    const int hi   = lane >> 4;     // K-segment selector (0 or 1)
    const int chunk  = Nn / NSPLIT; // 1024 rows per split (32 tiles of 32)
    const int n_base = s * chunk;
    const float* xb = x + (size_t)b * Nn * Cn;

    v8f a00 = {}, a01 = {}, a10 = {}, a11 = {};

#if HAS_TDM
    __shared__ float xs[2][32 * LDS_ROW];   // two padded 32x32 f32 tiles (8448 B)

    // Issue one TDM descriptor: 32x32 f32 tile at row n0 -> LDS buffer `buf`,
    // 1 pad dword after every 32 dwords (row stride LDS_ROW).
    auto tdm_issue = [&](int buf, int n0) {
        unsigned long long ga = (unsigned long long)(xb + (size_t)n0 * Cn);
        unsigned lds_off = (unsigned)(unsigned long long)(&xs[buf][0]); // low 32 bits = LDS offset
        v4u g0 = { 1u,                                   // count=1, user descriptor
                   lds_off,                              // D#.lds_addr
                   (unsigned)ga,                         // global_addr[31:0]
                   (unsigned)((ga >> 32) & 0x01FFFFFFu) | 0x80000000u }; // addr[56:32] | type=2
        v8i_ g1 = { (int)0x01120000,        // data_size=4B | pad_enable | pad_interval=32dw | pad=1dw
                    (int)(32u << 16),       // tensor_dim0 = 32 (low16 in [63:48])
                    (int)(32u << 16),       // tensor_dim1 = 32 (low16 in [95:80])
                    (int)(32u << 16),       // tile_dim0   = 32 ([127:112])
                    32,                     // tile_dim1   = 32 ([143:128]), tile_dim2 = 0
                    32,                     // tensor_dim0_stride = 32 ([191:160])
                    0, 0 };
        v4i_ gz4 = { 0, 0, 0, 0 };
        v8i_ gz8 = { 0, 0, 0, 0, 0, 0, 0, 0 };
        __builtin_amdgcn_tensor_load_to_lds(g0, g1, gz4, gz4, gz8, 0);
    };

    // Consume one staged tile: A from Ft (global bf16), B columns from LDS.
    auto step = [&](int buf, int kk) {
        const int n0 = n_base + kk + 16 * hi;
        v16bf af0 = *(const v16bf*)(Ft + (size_t)lo * Nn + n0);         // cos rows
        v16bf af1 = *(const v16bf*)(Ft + (size_t)(16 + lo) * Nn + n0);  // sin rows
        const float* ls = &xs[buf][(16 * hi) * LDS_ROW];
        v16bf bm0, bm1;
        #pragma unroll
        for (int k = 0; k < 16; ++k) {
            bm0[k] = (__bf16)ls[k * LDS_ROW + lo];
            bm1[k] = (__bf16)ls[k * LDS_ROW + 16 + lo];
        }
        a00 = __builtin_amdgcn_wmma_f32_16x16x32_bf16(false, af0, false, bm0, (short)0, a00, false, false);
        a01 = __builtin_amdgcn_wmma_f32_16x16x32_bf16(false, af0, false, bm1, (short)0, a01, false, false);
        a10 = __builtin_amdgcn_wmma_f32_16x16x32_bf16(false, af1, false, bm0, (short)0, a10, false, false);
        a11 = __builtin_amdgcn_wmma_f32_16x16x32_bf16(false, af1, false, bm1, (short)0, a11, false, false);
    };

    tdm_issue(0, n_base);
    tdm_issue(1, n_base + 32);
    for (int kk = 0; kk + 32 < chunk; kk += 32) {          // all but the last tile
        const int buf = (kk >> 5) & 1;
        __builtin_amdgcn_s_wait_tensorcnt(1);              // in-order: oldest tile landed
        step(buf, kk);
        if (kk + 64 < chunk) {
            asm volatile("s_wait_dscnt 0" ::: "memory");   // LDS reads of this buffer done
            tdm_issue(buf, n_base + kk + 64);              // refill for tile t+2
        }
    }
    __builtin_amdgcn_s_wait_tensorcnt(0);                  // last tile fully landed
    step(((chunk >> 5) - 1) & 1, chunk - 32);
#else
    for (int kk = 0; kk < chunk; kk += 32) {
        const int n0 = n_base + kk + 16 * hi;
        v16bf af0 = *(const v16bf*)(Ft + (size_t)lo * Nn + n0);
        v16bf af1 = *(const v16bf*)(Ft + (size_t)(16 + lo) * Nn + n0);
        v16bf bm0, bm1;
        #pragma unroll
        for (int k = 0; k < 16; ++k) {
            const float* row = xb + (size_t)(n0 + k) * Cn;
            bm0[k] = (__bf16)row[lo];
            bm1[k] = (__bf16)row[16 + lo];
        }
        a00 = __builtin_amdgcn_wmma_f32_16x16x32_bf16(false, af0, false, bm0, (short)0, a00, false, false);
        a01 = __builtin_amdgcn_wmma_f32_16x16x32_bf16(false, af0, false, bm1, (short)0, a01, false, false);
        a10 = __builtin_amdgcn_wmma_f32_16x16x32_bf16(false, af1, false, bm0, (short)0, a10, false, false);
        a11 = __builtin_amdgcn_wmma_f32_16x16x32_bf16(false, af1, false, bm1, (short)0, a11, false, false);
    }
#endif

    // D layout: VGPR v holds (M = v + 8*hi, N = lo)
    float* P = partial + (size_t)(b * NSPLIT + s) * (2 * MODESn * Cn);
    #pragma unroll
    for (int v = 0; v < 8; ++v) {
        const int jm = v + 8 * hi;
        P[jm * Cn + lo]              = a00[v];
        P[jm * Cn + 16 + lo]         = a01[v];
        P[(16 + jm) * Cn + lo]       = a10[v];
        P[(16 + jm) * Cn + 16 + lo]  = a11[v];
    }
}

// ---------------------------------------------------------------------------
// Kernel 2: reduce split-K partials, complex spectral mix, fold irfft scaling.
//   Re(xf[m,i]) = S[m][i], Im(xf[m,i]) = -S[16+m][i]
//   of = xf * (Wr + i*Wi); coef[m] = Re(of)*(m==0?1:2)/N; coef[16+m] = -Im(of)*(m==0?0:2)/N
// (irfft drops Im of the DC bin; modes 1..15 carry the Hermitian factor 2.)
// ---------------------------------------------------------------------------
__global__ void __launch_bounds__(512) fno_mix(const float* __restrict__ partial,
                                               const float* __restrict__ Wr,
                                               const float* __restrict__ Wi,
                                               float*       __restrict__ coef,
                                               int d) {
    __shared__ float S[2 * MODESn][Cn];
    const int b = blockIdx.x;
    const int t = threadIdx.x;
    for (int e = t; e < 2 * MODESn * Cn; e += 512) {
        float acc = 0.0f;
        const float* P = partial + (size_t)b * NSPLIT * (2 * MODESn * Cn) + e;
        #pragma unroll
        for (int sp = 0; sp < NSPLIT; ++sp) acc += P[(size_t)sp * (2 * MODESn * Cn)];
        S[e >> 5][e & 31] = acc;
    }
    __syncthreads();
    const int m = t >> 5;   // 0..15
    const int o = t & 31;   // 0..31
    const float* wr = Wr + (size_t)d * Cn * Cn * MODESn;
    const float* wi = Wi + (size_t)d * Cn * Cn * MODESn;
    float ofr = 0.0f, ofi = 0.0f;
    #pragma unroll
    for (int i = 0; i < Cn; ++i) {
        const float xr = S[m][i];
        const float xi = -S[16 + m][i];
        const float r  = wr[(i * Cn + o) * MODESn + m];
        const float im = wi[(i * Cn + o) * MODESn + m];
        ofr += xr * r - xi * im;
        ofi += xr * im + xi * r;
    }
    const float invN = 1.0f / (float)Nn;
    const float sc   = (m == 0 ? 1.0f : 2.0f) * invN;
    const float sc2  = (m == 0 ? 0.0f : 2.0f) * invN;
    float* cb = coef + (size_t)b * (2 * MODESn) * Cn;
    cb[m * Cn + o]        = ofr * sc;
    cb[(16 + m) * Cn + o] = -(ofi * sc2);
}

// ---------------------------------------------------------------------------
// Kernel 3: fused irfft + pointwise conv + bias + gelu, in place.
//   x_new = gelu( [F | x] (Nx64) * [coef; K] (64x32) + bias )
// B operands are M-tile-invariant: loaded once per wave, reused for 16 tiles.
// ---------------------------------------------------------------------------
__global__ void __launch_bounds__(32) fno_inverse_fused(const float*  __restrict__ xin,
                                                        float*        __restrict__ xout,
                                                        const __bf16* __restrict__ F,
                                                        const float*  __restrict__ coef,
                                                        const float*  __restrict__ convk,
                                                        const float*  __restrict__ convb,
                                                        int d) {
    const int b     = blockIdx.x;
    const int wslot = blockIdx.y;                  // 0..63, 16 tiles each
    const int lane  = threadIdx.x;
    const int lo    = lane & 15;
    const int hi    = lane >> 4;

    const float* cb = coef  + (size_t)b * (2 * MODESn) * Cn;
    const float* kd = convk + (size_t)d * Cn * Cn;
    v16bf Bc0, Bc1, Bk0, Bk1;                      // tile-invariant B operands
    #pragma unroll
    for (int k = 0; k < 16; ++k) {
        const int row = 16 * hi + k;
        Bc0[k] = (__bf16)cb[row * Cn + lo];
        Bc1[k] = (__bf16)cb[row * Cn + 16 + lo];
        Bk0[k] = (__bf16)kd[row * Cn + lo];
        Bk1[k] = (__bf16)kd[row * Cn + 16 + lo];
    }
    const float bias0 = convb[d * Cn + lo];
    const float bias1 = convb[d * Cn + 16 + lo];

    const float* xi = xin  + (size_t)b * Nn * Cn;
    float*       xo = xout + (size_t)b * Nn * Cn;

    for (int it = 0; it < TILES_PER_WAVE; ++it) {
        const int n_base = (wslot * TILES_PER_WAVE + it) * 16;
        // A kstep0: basis rows (bf16, contiguous)
        v16bf aF = *(const v16bf*)(F + (size_t)(n_base + lo) * (2 * MODESn) + 16 * hi);
        // A kstep1: x rows, f32 -> bf16 (4x b128 loads, 16 full rows per wave)
        const float4* xr4 = (const float4*)(xi + (size_t)(n_base + lo) * Cn + 16 * hi);
        float4 f0 = xr4[0], f1 = xr4[1], f2 = xr4[2], f3 = xr4[3];
        const float fv[16] = { f0.x, f0.y, f0.z, f0.w, f1.x, f1.y, f1.z, f1.w,
                               f2.x, f2.y, f2.z, f2.w, f3.x, f3.y, f3.z, f3.w };
        v16bf aX;
        #pragma unroll
        for (int k = 0; k < 16; ++k) aX[k] = (__bf16)fv[k];

        if (it + 1 < TILES_PER_WAVE)  // feed next tile's x into cache (global_prefetch path)
            __builtin_prefetch(xi + (size_t)(n_base + 16 + lo) * Cn + 16 * hi, 0, 1);

        v8f acc0 = {}, acc1 = {};
        acc0 = __builtin_amdgcn_wmma_f32_16x16x32_bf16(false, aF, false, Bc0, (short)0, acc0, false, false);
        acc0 = __builtin_amdgcn_wmma_f32_16x16x32_bf16(false, aX, false, Bk0, (short)0, acc0, false, false);
        acc1 = __builtin_amdgcn_wmma_f32_16x16x32_bf16(false, aF, false, Bc1, (short)0, acc1, false, false);
        acc1 = __builtin_amdgcn_wmma_f32_16x16x32_bf16(false, aX, false, Bk1, (short)0, acc1, false, false);

        #pragma unroll
        for (int v = 0; v < 8; ++v) {
            const int n = n_base + v + 8 * hi;     // D: (M = v + 8*hi, N = lo)
            xo[(size_t)n * Cn + lo]      = gelu_tanh(acc0[v] + bias0);
            xo[(size_t)n * Cn + 16 + lo] = gelu_tanh(acc1[v] + bias1);
        }
    }
}

// ---------------------------------------------------------------------------
// ws layout: [0,1MB) Ft bf16 | [1MB,2MB) F bf16 | [2MB,6MB) partials f32
//            [6MB,6.25MB) coef f32   (total ~6.3 MB)
// x ping-pongs entirely inside d_out (depth 0 reads d_in; all depths write d_out
// in place; per-wave row ranges are disjoint and loads complete before stores).
// ---------------------------------------------------------------------------
extern "C" void kernel_launch(void* const* d_in, const int* in_sizes, int n_in,
                              void* d_out, int out_size, void* d_ws, size_t ws_size,
                              hipStream_t stream) {
    const float* x0    = (const float*)d_in[0];
    const float* Wr    = (const float*)d_in[1];
    const float* Wi    = (const float*)d_in[2];
    const float* convk = (const float*)d_in[3];
    const float* convb = (const float*)d_in[4];
    float* xio = (float*)d_out;

    char*   ws      = (char*)d_ws;
    __bf16* Ft      = (__bf16*)(ws);
    __bf16* F       = (__bf16*)(ws + (1u << 20));
    float*  partial = (float*)(ws + (2u << 20));
    float*  coef    = (float*)(ws + (6u << 20));

    build_basis<<<dim3(Nn / 256), dim3(256), 0, stream>>>(Ft, F);
    for (int d = 0; d < DEPTHn; ++d) {
        const float* xin = (d == 0) ? x0 : xio;
        fno_project<<<dim3(Bn, NSPLIT), dim3(32), 0, stream>>>(xin, Ft, partial);
        fno_mix<<<dim3(Bn), dim3(512), 0, stream>>>(partial, Wr, Wi, coef, d);
        fno_inverse_fused<<<dim3(Bn, Nn / (16 * TILES_PER_WAVE)), dim3(32), 0, stream>>>(
            xin, xio, F, coef, convk, convb, d);
    }
}
